// Net_90744069030478
// MI455X (gfx1250) — matrix operand order repaired and checked by reading, not compiled, not run on hardware
//
#include <hip/hip_runtime.h>
#include <hip/hip_bf16.h>
#include <math.h>

#define NN 100000
#define FF 500
#define HH 64
#define CC 40
#define EE 1000000
#define KP1 512   // FF padded to multiple of 32
#define K23 192   // 3*HH, already a multiple of 32

typedef __attribute__((ext_vector_type(16))) __bf16 v16bf;
typedef __attribute__((ext_vector_type(8)))  __bf16 v8bf;
typedef __attribute__((ext_vector_type(8)))  float  v8f;

// ---------------------------------------------------------------------------
// x (f32 N x FF) -> zero-padded bf16 N x KP1, row major. 4 elems/thread.
// ---------------------------------------------------------------------------
__global__ void convert_x(const float* __restrict__ x, __bf16* __restrict__ xb) {
  size_t base = ((size_t)blockIdx.x * blockDim.x + threadIdx.x) * 4;
  if (base >= (size_t)NN * KP1) return;
  int row = (int)(base >> 9);          // /512
  int k0  = (int)(base & 511);
  const float* xr = x + (size_t)row * FF;
#pragma unroll
  for (int j = 0; j < 4; ++j) {
    int k = k0 + j;
    xb[base + j] = (__bf16)((k < FF) ? xr[k] : 0.0f);
  }
}

// ---------------------------------------------------------------------------
// u (f32 N x 192) -> bf16( relu(u) * filt_concat ), row major. 4 elems/thread.
// ---------------------------------------------------------------------------
__global__ void convert_relu_filt(const float* __restrict__ u, __bf16* __restrict__ ub,
                                  const float* __restrict__ f0,
                                  const float* __restrict__ f1,
                                  const float* __restrict__ f2) {
  size_t base = ((size_t)blockIdx.x * blockDim.x + threadIdx.x) * 4;
  if (base >= (size_t)NN * K23) return;
  int k0 = (int)(base % K23);                  // 4-aligned, never crosses a 64-block
  const float* fp = (k0 < 64) ? f0 : ((k0 < 128) ? f1 : f2);
  const float4 v = *(const float4*)(u + base);
  float r0 = fmaxf(v.x, 0.0f) * fp[(k0 + 0) & 63];
  float r1 = fmaxf(v.y, 0.0f) * fp[(k0 + 1) & 63];
  float r2 = fmaxf(v.z, 0.0f) * fp[(k0 + 2) & 63];
  float r3 = fmaxf(v.w, 0.0f) * fp[(k0 + 3) & 63];
  ub[base + 0] = (__bf16)r0;
  ub[base + 1] = (__bf16)r1;
  ub[base + 2] = (__bf16)r2;
  ub[base + 3] = (__bf16)r3;
}

// ---------------------------------------------------------------------------
// Pack W (f32 K x ncols) into WMMA B-fragment order, zero-padded:
// Wp[frag=kt32*NT+t][lane][i] = bf16(W[k][n]); k = kt32*32 + (i<8 ? 8*(lane>>4)+i : 16+8*(lane>>4)+i-8),
// n = t*16 + (lane&15). One thread per (frag, lane) -> writes 16 bf16 (32B).
// ---------------------------------------------------------------------------
template <int KPAD, int NT>
__global__ void pack_w(const float* __restrict__ W, int K, int ncols,
                       __bf16* __restrict__ Wp) {
  int gid = blockIdx.x * blockDim.x + threadIdx.x;
  const int total = (KPAD / 32) * NT * 32;
  if (gid >= total) return;
  int lane  = gid & 31;
  int frag  = gid >> 5;
  int t     = frag % NT;
  int kt32  = frag / NT;
  int n     = t * 16 + (lane & 15);
  int khalf = lane >> 4;
  __bf16* dst = Wp + (size_t)gid * 16;
#pragma unroll
  for (int i = 0; i < 16; ++i) {
    int k = kt32 * 32 + ((i < 8) ? (8 * khalf + i) : (16 + 8 * khalf + (i - 8)));
    float v = (k < K && n < ncols) ? W[(size_t)k * ncols + n] : 0.0f;
    dst[i] = (__bf16)v;
  }
}

// ---------------------------------------------------------------------------
// Guard-free bf16 WMMA GEMM: Out[M,ncols] = A[M,KPAD]bf16 @ Wp + bias.
// One wave per 16-row tile, NT 16-column accumulators (V_WMMA_F32_16X16X32_BF16).
// A fragment = two aligned b128 loads; B fragment = two lane-contiguous b128 loads.
// ---------------------------------------------------------------------------
template <int KPAD, int NT>
__global__ void gemm_bf16(const __bf16* __restrict__ A, const __bf16* __restrict__ Wp,
                          const float* __restrict__ bias, float* __restrict__ Out,
                          int M, int ncols) {
  const int lane  = threadIdx.x & 31;
  const int wave  = threadIdx.x >> 5;
  const int rowT  = blockIdx.x * 4 + wave;
  if (rowT * 16 >= M) return;
  const int row0  = rowT * 16;
  const int khalf = lane >> 4;
  const int row   = row0 + (lane & 15);

  const __bf16* Ar = A + (size_t)row * KPAD + 8 * khalf;
  const v16bf*  Wf = (const v16bf*)Wp;

  v8f acc[NT] = {};

  for (int kt32 = 0; kt32 < KPAD / 32; ++kt32) {
    const v8bf c0 = *(const v8bf*)(Ar + kt32 * 32);
    const v8bf c1 = *(const v8bf*)(Ar + kt32 * 32 + 16);
    v16bf a;
#pragma unroll
    for (int i = 0; i < 8; ++i) { a[i] = c0[i]; a[i + 8] = c1[i]; }
#pragma unroll
    for (int t = 0; t < NT; ++t) {
      v16bf b = Wf[(size_t)(kt32 * NT + t) * 32 + lane];
      acc[t] = __builtin_amdgcn_wmma_f32_16x16x32_bf16(
          false, a, false, b, (short)0, acc[t], false, false);
    }
  }

#pragma unroll
  for (int t = 0; t < NT; ++t) {
    int n = t * 16 + (lane & 15);
    if (n < ncols) {
      float bv = bias[n];
#pragma unroll
      for (int r = 0; r < 8; ++r)
        Out[(size_t)(row0 + r + 8 * khalf) * ncols + n] = acc[t][r] + bv;
    }
  }
}

// ---------------------------------------------------------------------------
// Edge scatter: out[dst, col0+f] += attr_e * h[src, f]. 8 threads/edge,
// two float4 gathers + 8 f32 atomics; h and out are L2-resident (192 MB L2).
// ---------------------------------------------------------------------------
__global__ void ufg_scatter(const float* __restrict__ h, const int* __restrict__ ei,
                            const float* __restrict__ ea, float* __restrict__ out,
                            int col0) {
  int gid = blockIdx.x * blockDim.x + threadIdx.x;
  int e   = gid >> 3;
  if (e >= EE) return;
  int f0  = (gid & 7) * 8;
  int src = ei[e];
  int dst = ei[EE + e];
  float a = ea[e];
  const float4* hv = (const float4*)(h + (size_t)src * HH + f0);
  float4 v0 = hv[0];
  float4 v1 = hv[1];
  float* o = out + (size_t)dst * K23 + col0 + f0;
  atomicAdd(o + 0, a * v0.x);
  atomicAdd(o + 1, a * v0.y);
  atomicAdd(o + 2, a * v0.z);
  atomicAdd(o + 3, a * v0.w);
  atomicAdd(o + 4, a * v1.x);
  atomicAdd(o + 5, a * v1.y);
  atomicAdd(o + 6, a * v1.z);
  atomicAdd(o + 7, a * v1.w);
}

// ---------------------------------------------------------------------------
// log_softmax over 40 classes, one wave per row (wave32 shfl_xor reductions).
// ---------------------------------------------------------------------------
__global__ void log_softmax_rows(const float* __restrict__ logits, float* __restrict__ out) {
  int lane = threadIdx.x & 31;
  int wave = threadIdx.x >> 5;
  int row  = blockIdx.x * 4 + wave;
  if (row >= NN) return;
  const float* L = logits + (size_t)row * CC;
  float v0 = (lane < CC) ? L[lane] : -1e30f;
  float v1 = (lane + 32 < CC) ? L[lane + 32] : -1e30f;
  float mx = fmaxf(v0, v1);
#pragma unroll
  for (int off = 16; off > 0; off >>= 1) mx = fmaxf(mx, __shfl_xor(mx, off, 32));
  float s = ((lane < CC) ? expf(v0 - mx) : 0.0f) +
            ((lane + 32 < CC) ? expf(v1 - mx) : 0.0f);
#pragma unroll
  for (int off = 16; off > 0; off >>= 1) s += __shfl_xor(s, off, 32);
  float lg = logf(s);
  if (lane < CC)      out[(size_t)row * CC + lane]      = v0 - mx - lg;
  if (lane + 32 < CC) out[(size_t)row * CC + lane + 32] = v1 - mx - lg;
}

// ---------------------------------------------------------------------------
extern "C" void kernel_launch(void* const* d_in, const int* in_sizes, int n_in,
                              void* d_out, int out_size, void* d_ws, size_t ws_size,
                              hipStream_t stream) {
  (void)in_sizes; (void)n_in; (void)out_size; (void)ws_size;

  const float* x    = (const float*)d_in[0];
  const int*   ei1  = (const int*)d_in[1];
  const float* ea1  = (const float*)d_in[2];
  const int*   ei2  = (const int*)d_in[3];
  const float* ea2  = (const float*)d_in[4];
  const int*   ei3  = (const int*)d_in[5];
  const float* ea3  = (const float*)d_in[6];
  const float* W1   = (const float*)d_in[7];
  const float* b1   = (const float*)d_in[8];
  const float* f1_1 = (const float*)d_in[9];
  const float* f2_1 = (const float*)d_in[10];
  const float* f1_2 = (const float*)d_in[11];
  const float* f2_2 = (const float*)d_in[12];
  const float* f1_3 = (const float*)d_in[13];
  const float* f2_3 = (const float*)d_in[14];
  const float* Wm2  = (const float*)d_in[15];
  const float* bm2  = (const float*)d_in[16];
  const float* Wm1  = (const float*)d_in[17];
  const float* bm1  = (const float*)d_in[18];
  float* out = (float*)d_out;

  // ---- workspace layout (bytes; all offsets 256B-aligned) -----------------
  char* ws = (char*)d_ws;
  __bf16* xb     = (__bf16*)ws;                                   // N*512*2 = 102,400,000
  __bf16* ub     = (__bf16*)ws;                                   // reuses xb region after gemm1 (N*192*2)
  __bf16* w1p    = (__bf16*)(ws + 102400000);                     // 2048 frags * 32B = 65,536
  __bf16* wm2p   = (__bf16*)(ws + 102400000 + 65536);             // 768 frags  * 32B = 24,576
  __bf16* wm1p   = (__bf16*)(ws + 102400000 + 65536 + 24576);     // 576 frags  * 32B = 18,432
  float*  h      = (float*)(ws + 102508544);                      // N*64*4  = 25,600,000
  float*  u      = (float*)(ws + 128108544);                      // N*192*4 = 76,800,000
  float*  logits = (float*)(ws + 204908544);                      // N*40*4  = 16,000,000

  const int gemmBlocks    = (NN / 16 + 3) / 4;
  const int scatterBlocks = (EE * 8 + 255) / 256;
  const int smBlocks      = (NN + 3) / 4;

  // ---- stage 0: conversions / packing ------------------------------------
  convert_x<<<(int)(((size_t)NN * KP1 / 4 + 255) / 256), 256, 0, stream>>>(x, xb);
  pack_w<KP1, 4><<<8, 256, 0, stream>>>(W1, FF, HH, w1p);
  pack_w<K23, 4><<<3, 256, 0, stream>>>(Wm2, K23, HH, wm2p);
  pack_w<K23, 3><<<3, 256, 0, stream>>>(Wm1, K23, CC, wm1p);

  // ---- h = x @ W1 + b1 ----------------------------------------------------
  gemm_bf16<KP1, 4><<<gemmBlocks, 128, 0, stream>>>(xb, w1p, b1, h, NN, HH);

  // ---- block 1 UFG scatters ----------------------------------------------
  hipMemsetAsync(u, 0, (size_t)NN * K23 * sizeof(float), stream);
  ufg_scatter<<<scatterBlocks, 256, 0, stream>>>(h, ei1, ea1, u, 0);
  ufg_scatter<<<scatterBlocks, 256, 0, stream>>>(h, ei2, ea2, u, 64);
  ufg_scatter<<<scatterBlocks, 256, 0, stream>>>(h, ei3, ea3, u, 128);

  // ---- h = relu(u * filt1) @ W_mlp2 + b_mlp2 ------------------------------
  convert_relu_filt<<<(int)(((size_t)NN * K23 / 4 + 255) / 256), 256, 0, stream>>>(
      u, ub, f1_1, f1_2, f1_3);
  gemm_bf16<K23, 4><<<gemmBlocks, 128, 0, stream>>>(ub, wm2p, bm2, h, NN, HH);

  // ---- block 2 UFG scatters ----------------------------------------------
  hipMemsetAsync(u, 0, (size_t)NN * K23 * sizeof(float), stream);
  ufg_scatter<<<scatterBlocks, 256, 0, stream>>>(h, ei1, ea1, u, 0);
  ufg_scatter<<<scatterBlocks, 256, 0, stream>>>(h, ei2, ea2, u, 64);
  ufg_scatter<<<scatterBlocks, 256, 0, stream>>>(h, ei3, ea3, u, 128);

  // ---- logits = relu(u * filt2) @ W_mlp1 + b_mlp1 -------------------------
  convert_relu_filt<<<(int)(((size_t)NN * K23 / 4 + 255) / 256), 256, 0, stream>>>(
      u, ub, f2_1, f2_2, f2_3);
  gemm_bf16<K23, 3><<<gemmBlocks, 128, 0, stream>>>(ub, wm1p, bm1, logits, NN, CC);

  // ---- out = log_softmax(logits) -----------------------------------------
  log_softmax_rows<<<smBlocks, 128, 0, stream>>>(logits, out);
}
